// AttentionHead_23991687315790
// MI455X (gfx1250) — matrix operand order repaired and checked by reading, not compiled, not run on hardware
//
#include <hip/hip_runtime.h>

// B=8, S=2048, DIN=1024, DQ=DK=128.
// Workspace layout (needs ~13.3 MiB):
//   [0)                Wfrag   : 3*256 tiles * 32 lanes * 8 dwords  = 786432 B
//   [786432)           Qb bf16 : 16384*128*2                        = 4 MiB
//   [+4MiB)            KT bf16 : [8][128][2048] (K transposed)      = 4 MiB
//   [+8MiB)            VT bf16 : [8][128][2048] (V transposed)      = 4 MiB
//   [+12MiB)           KtV f32 : [8][128][128]                      = 512 KiB

typedef __attribute__((ext_vector_type(16))) __bf16 v16bf;
typedef __attribute__((ext_vector_type(8)))  float  v8f;

struct alignas(16) U4 { unsigned x, y, z, w; };

union Frag16 {
  v16bf v;
  __bf16 b[16];
  unsigned short h[16];
  unsigned u[8];
};

// Native bf16 convert: let clang pick v_cvt_*_bf16_f32 on gfx1250.
__device__ __forceinline__ __bf16 cvt_bf16(float f) { return (__bf16)f; }

__device__ __forceinline__ unsigned short bf_bits(float f) {
  return __builtin_bit_cast(unsigned short, (__bf16)f);
}

__device__ __forceinline__ v16bf frag_from_u4(U4 a, U4 b) {
  Frag16 f;
  f.u[0] = a.x; f.u[1] = a.y; f.u[2] = a.z; f.u[3] = a.w;
  f.u[4] = b.x; f.u[5] = b.y; f.u[6] = b.z; f.u[7] = b.w;
  return f.v;
}

__device__ __forceinline__ v8f wmma_bf16(v16bf a, v16bf b, v8f c) {
  // (neg_a, A, neg_b, B, c_mod, C, reuse_a, reuse_b)
  return __builtin_amdgcn_wmma_f32_16x16x32_bf16(false, a, false, b, (short)0, c,
                                                 false, false);
}

// A fragment 16x32 from row-major f32 [., ld]; contiguous k order (used by A and B alike):
// lanes 0-15: K 0..15; lanes 16-31: K 16..31; M = lane%16.
__device__ __forceinline__ v16bf load_a_f32(const float* X, int ld, int row0,
                                            int k0, int lane) {
  const int m   = lane & 15;
  const int off = (lane >> 4) << 4;  // 0 or 16
  const float* p = X + (size_t)(row0 + m) * ld + (k0 + off);
  Frag16 f;
#pragma unroll
  for (int i = 0; i < 8; ++i) f.b[i] = cvt_bf16(p[i]);
#pragma unroll
  for (int i = 0; i < 8; ++i) f.b[8 + i] = cvt_bf16(p[8 + i]);
  return f.v;
}

// Same A fragment from row-major bf16 (two 16B vector loads per lane).
__device__ __forceinline__ v16bf load_a_bf16(const unsigned short* X, int ld,
                                             int row0, int k0, int lane) {
  const int m   = lane & 15;
  const int off = (lane >> 4) << 4;
  const unsigned short* p = X + (size_t)(row0 + m) * ld + (k0 + off);
  U4 a = *(const U4*)(p);
  U4 b = *(const U4*)(p + 8);
  return frag_from_u4(a, b);
}

// B fragment 32x16 where B[k][n] = BT[(n0+n)*ld + k] (BT = transposed bf16).
// lanes 0-15: K 0..15, lanes 16-31: K 16..31; N = lane%16 -> contiguous 32B.
__device__ __forceinline__ v16bf load_b_bt(const unsigned short* BT, int ld,
                                           int k0, int n0, int lane) {
  const int n   = lane & 15;
  const int off = (lane >> 4) << 4;  // 0 or 16
  const unsigned short* p = BT + (size_t)(n0 + n) * ld + (k0 + off);
  U4 a = *(const U4*)(p);
  U4 b = *(const U4*)(p + 8);
  return frag_from_u4(a, b);
}

// ---------------- Kernel 0: pre-swizzle weights into B-fragment layout ------
__global__ void prep_wfrag_kernel(const float* __restrict__ Wq,
                                  const float* __restrict__ Wk,
                                  const float* __restrict__ Wv,
                                  unsigned* __restrict__ wfrag) {
  const int lane = threadIdx.x;  // 32 threads
  const int bid  = blockIdx.x;   // mat*256 + kt*8 + nt  (768 blocks)
  const int mat  = bid >> 8;
  const int rem  = bid & 255;
  const int kt   = rem >> 3;
  const int nt   = rem & 7;
  const float* W = (mat == 0) ? Wq : (mat == 1) ? Wk : Wv;
  const int n     = nt * 16 + (lane & 15);
  const int kbase = kt * 32 + ((lane >> 4) << 4);
  Frag16 f;
#pragma unroll
  for (int i = 0; i < 16; ++i)
    f.b[i] = cvt_bf16(W[(size_t)(kbase + i) * 128 + n]);
  unsigned* dst = wfrag + ((size_t)bid * 32 + lane) * 8;
#pragma unroll
  for (int i = 0; i < 8; ++i) dst[i] = f.u[i];
}

// ---------------- Kernel 1: Q/K/V projections -------------------------------
__global__ void __launch_bounds__(256)
proj_kernel(const float* __restrict__ Xq, const float* __restrict__ Xk,
            const float* __restrict__ Xv, const unsigned* __restrict__ wfrag,
            const float* __restrict__ bq, const float* __restrict__ bk,
            const float* __restrict__ bv, unsigned short* __restrict__ Qb,
            unsigned short* __restrict__ KT, unsigned short* __restrict__ VT) {
  const int mat = blockIdx.y;  // 0=Q 1=K 2=V
  const float* X    = (mat == 0) ? Xq : (mat == 1) ? Xk : Xv;
  const float* bias = (mat == 0) ? bq : (mat == 1) ? bk : bv;
  const unsigned* frg = wfrag + (size_t)mat * 256 * 32 * 8;

  const int lane = threadIdx.x & 31;
  const int wave = threadIdx.x >> 5;
  const int row0 = blockIdx.x * 128 + wave * 16;

  v8f acc[8];
#pragma unroll
  for (int nt = 0; nt < 8; ++nt)
    acc[nt] = (v8f){0.f, 0.f, 0.f, 0.f, 0.f, 0.f, 0.f, 0.f};

  const int m_l   = lane & 15;
  const int a_off = (lane >> 4) << 3;
  const float* arow = X + (size_t)(row0 + m_l) * 1024 + a_off;

  for (int kt = 0; kt < 32; ++kt) {
    // Prefetch next K-tile of the activation stream (global_prefetch_b8).
    if (kt + 1 < 32) __builtin_prefetch(arow + (kt + 1) * 32, 0, 1);
    v16bf a = load_a_f32(X, 1024, row0, kt * 32, lane);
    const unsigned* fb = frg + ((size_t)(kt * 8) * 32 + lane) * 8;
#pragma unroll
    for (int nt = 0; nt < 8; ++nt) {
      const U4* p = (const U4*)(fb + (size_t)nt * 32 * 8);
      v16bf b = frag_from_u4(p[0], p[1]);
      acc[nt] = wmma_bf16(a, b, acc[nt]);
    }
  }

  const int n_l = lane & 15;
  const int hi  = lane >> 4;
  if (mat == 0) {
    // Q: row-major bf16 [16384][128]
#pragma unroll
    for (int nt = 0; nt < 8; ++nt) {
      const int col = nt * 16 + n_l;
      const float bb = bias[col];
#pragma unroll
      for (int j = 0; j < 8; ++j) {
        const int r = row0 + j + 8 * hi;
        Qb[(size_t)r * 128 + col] = bf_bits(acc[nt][j] + bb);
      }
    }
  } else {
    // K/V: transposed bf16 [8][128][2048]; lane's 8 values are consecutive s.
    unsigned short* T = (mat == 1) ? KT : VT;
    const int b  = row0 >> 11;
    const int sb = (row0 & 2047) + 8 * hi;
#pragma unroll
    for (int nt = 0; nt < 8; ++nt) {
      const int col = nt * 16 + n_l;
      const float bb = bias[col];
      unsigned hs[8];
#pragma unroll
      for (int j = 0; j < 8; ++j) hs[j] = bf_bits(acc[nt][j] + bb);
      U4 o;
      o.x = hs[0] | (hs[1] << 16);
      o.y = hs[2] | (hs[3] << 16);
      o.z = hs[4] | (hs[5] << 16);
      o.w = hs[6] | (hs[7] << 16);
      *(U4*)(T + ((size_t)b * 128 + col) * 2048 + sb) = o;
    }
  }
}

// ---------------- Kernel 2: zero the KtV accumulator ------------------------
__global__ void zero_f32_kernel(float* __restrict__ p, int n) {
  int i = blockIdx.x * blockDim.x + threadIdx.x;
  if (i < n) p[i] = 0.0f;
}

// ---------------- Kernel 3: KtV[b] = K^T @ V (split-K + f32 atomics) --------
__global__ void __launch_bounds__(256)
ktv_kernel(const unsigned short* __restrict__ KT,
           const unsigned short* __restrict__ VT, float* __restrict__ KtV) {
  const int split = blockIdx.x;  // 16 splits over S
  const int b     = blockIdx.y;  // 8 batches
  const int lane  = threadIdx.x & 31;
  const int wave  = threadIdx.x >> 5;
  const int m0    = wave * 16;
  const unsigned short* KTb = KT + (size_t)b * 128 * 2048;
  const unsigned short* VTb = VT + (size_t)b * 128 * 2048;

  v8f acc[8];
#pragma unroll
  for (int nt = 0; nt < 8; ++nt)
    acc[nt] = (v8f){0.f, 0.f, 0.f, 0.f, 0.f, 0.f, 0.f, 0.f};

  for (int it = 0; it < 4; ++it) {
    const int s0 = split * 128 + it * 32;
    v16bf a = load_a_bf16(KTb, 2048, m0, s0, lane);  // A = K^T (row-major KT)
#pragma unroll
    for (int nt = 0; nt < 8; ++nt) {
      v16bf bfr = load_b_bt(VTb, 2048, s0, nt * 16, lane);  // B = V
      acc[nt] = wmma_bf16(a, bfr, acc[nt]);
    }
  }

  const int n_l = lane & 15;
  const int hi  = lane >> 4;
  float* Cb = KtV + (size_t)b * 128 * 128;
#pragma unroll
  for (int nt = 0; nt < 8; ++nt) {
    const int col = nt * 16 + n_l;
#pragma unroll
    for (int j = 0; j < 8; ++j)
      atomicAdd(&Cb[(size_t)(m0 + j + 8 * hi) * 128 + col], acc[nt][j]);
  }
}

// ---------------- Kernel 4: out = (Q*scale) @ KtV[b] ------------------------
__global__ void __launch_bounds__(256)
out_kernel(const unsigned short* __restrict__ Qb, const float* __restrict__ KtV,
           float* __restrict__ out) {
  __shared__ __align__(16) unsigned short LT[128 * 128];  // KtV[b]^T in bf16

  const int tid  = threadIdx.x;
  const int b    = blockIdx.x >> 4;  // 16 blocks per batch (128 rows each)
  const float* Cb = KtV + (size_t)b * 16384;
  for (int t = 0; t < 64; ++t) {
    const int e = t * 256 + tid;
    const int k = e >> 7, n = e & 127;
    LT[n * 128 + k] = bf_bits(Cb[e]);
  }
  __syncthreads();

  const int lane = tid & 31;
  const int wave = tid >> 5;
  const int row0 = blockIdx.x * 128 + wave * 16;  // global Q row

  v8f acc[8];
#pragma unroll
  for (int nt = 0; nt < 8; ++nt)
    acc[nt] = (v8f){0.f, 0.f, 0.f, 0.f, 0.f, 0.f, 0.f, 0.f};

#pragma unroll
  for (int kt = 0; kt < 4; ++kt) {
    v16bf a = load_a_bf16(Qb, 128, row0, kt * 32, lane);
#pragma unroll
    for (int nt = 0; nt < 8; ++nt) {
      const int nn  = nt * 16 + (lane & 15);
      const int off = (lane >> 4) << 4;
      const unsigned short* p = &LT[nn * 128 + kt * 32 + off];
      U4 x = *(const U4*)p;
      U4 y = *(const U4*)(p + 8);
      acc[nt] = wmma_bf16(a, frag_from_u4(x, y), acc[nt]);
    }
  }

  const float scale = 0.08838834764831845f;  // 1/sqrt(128)
  const int n_l = lane & 15;
  const int hi  = lane >> 4;
#pragma unroll
  for (int nt = 0; nt < 8; ++nt) {
    const int col = nt * 16 + n_l;
#pragma unroll
    for (int j = 0; j < 8; ++j)
      out[(size_t)(row0 + j + 8 * hi) * 128 + col] = acc[nt][j] * scale;
  }
}

extern "C" void kernel_launch(void* const* d_in, const int* in_sizes, int n_in,
                              void* d_out, int out_size, void* d_ws,
                              size_t ws_size, hipStream_t stream) {
  const float* query = (const float*)d_in[0];
  const float* key_  = (const float*)d_in[1];
  const float* value = (const float*)d_in[2];
  const float* Wq    = (const float*)d_in[3];
  const float* bq    = (const float*)d_in[4];
  const float* Wk    = (const float*)d_in[5];
  const float* bk    = (const float*)d_in[6];
  const float* Wv    = (const float*)d_in[7];
  const float* bv    = (const float*)d_in[8];
  float* out = (float*)d_out;

  char* ws = (char*)d_ws;
  const size_t WFRAG_B = 786432;          // 3*256*32*8*4
  const size_t MAT_B   = 4194304;         // 16384*128*2
  unsigned*       wfrag = (unsigned*)ws;
  unsigned short* Qb    = (unsigned short*)(ws + WFRAG_B);
  unsigned short* KT    = (unsigned short*)(ws + WFRAG_B + MAT_B);
  unsigned short* VT    = (unsigned short*)(ws + WFRAG_B + 2 * MAT_B);
  float*          KtV   = (float*)(ws + WFRAG_B + 3 * MAT_B);  // 512 KiB

  prep_wfrag_kernel<<<768, 32, 0, stream>>>(Wq, Wk, Wv, wfrag);
  proj_kernel<<<dim3(128, 3), 256, 0, stream>>>(query, key_, value, wfrag, bq,
                                                bk, bv, Qb, KT, VT);
  zero_f32_kernel<<<512, 256, 0, stream>>>(KtV, 8 * 128 * 128);
  ktv_kernel<<<dim3(16, 8), 256, 0, stream>>>(KT, VT, KtV);
  out_kernel<<<128, 256, 0, stream>>>(Qb, KtV, out);
}
